// CustomLSTM_75917841924156
// MI455X (gfx1250) — matrix operand order repaired and checked by reading, not compiled, not run on hardware
//
#include <hip/hip_runtime.h>

// ---------------------------------------------------------------------------
// 4-layer LSTM (B=4096, IN=H=1024) + FC on CDNA5 / gfx1250 (wave32, WMMA).
//
// Compute-bound (~650 FLOP/byte; 67 MB bf16 weights sit in the 192 MB L2).
//   1) One-time repack of weights: fp32 -> bf16 in WMMA B-fragment order
//      (32x16 tiles, 32 contiguous bytes/lane) -> B fragment = 2x b128 loads.
//   2) Per-layer repack of activations: fp32 -> bf16 in WMMA A-fragment order
//      -> A fragment = 2x b128 loads, zero cvt in the hot loop.
//   3) GEMM: V_WMMA_F32_16X16X32_BF16, f32 accumulate. Wave tile 32x16 x 4
//      gates (8 accumulators); B fragments reused by 2 WMMAs; LSTM cell
//      update is pure per-lane VALU (gate accumulators are element-aligned).
// Hot loop per K-step/wave: 12x global_load_b128 + 8x v_wmma. Fallback tiers
// (weights-only pack / raw gather) if ws_size is too small.
// ---------------------------------------------------------------------------

typedef __attribute__((ext_vector_type(16))) __bf16 v16bf;
typedef __attribute__((ext_vector_type(8)))  float  v8f;

#define LAYERS   4
#define BATCH    4096
#define HDIM     1024
#define GSTRIDE  4096            // 4*H, row stride of Wi/Wh
#define KTILE    32
#define KBLKS    (HDIM / KTILE)  // 32 K-blocks per 1024-deep matrix
#define FRAG_ELEMS 512           // bf16 elements per packed 16x32/32x16 frag

__device__ __forceinline__ float fast_sigmoid(float x) {
  return 1.0f / (1.0f + __expf(-x));
}
__device__ __forceinline__ float fast_tanh(float x) {
  return 2.0f / (1.0f + __expf(-2.0f * x)) - 1.0f;
}

// Packed fragment load: 512 contiguous bf16, lane's 16 elems = 32 bytes.
__device__ __forceinline__ v16bf load_frag_packed(const __bf16* __restrict__ p,
                                                  size_t frag, int lane) {
  return *(const v16bf*)(p + (frag << 9) + lane * 16);
}

// ---- A fragment, on-the-fly path (fp32 row-major activations).
// A 16x32 layout (matches B): lanes 0-15 -> M=lane, elems 0..15 = K 0..15;
// lanes 16-31 same M, K 16..31.
__device__ __forceinline__ v16bf load_a_frag(const float* __restrict__ arow,
                                             int kbase, int hi) {
  v16bf a;
#pragma unroll
  for (int p = 0; p < 8; ++p) {
    const int kk = kbase + 2 * p + hi * 16;
    const float2 f = *(const float2*)(arow + kk);
    a[2 * p]     = (__bf16)f.x;
    a[2 * p + 1] = (__bf16)f.y;
  }
  return a;
}

// ---- B fragment, gather fallback (fp32 strided source).
// B 32x16 layout as used here (contiguous k, same as A): lane = column; lanes 0-15 K 0..15, 16-31 K 16..31.
__device__ __forceinline__ v16bf load_b_gather(const float* __restrict__ wcol,
                                               int kbase, int hi, int wstride) {
  v16bf b;
#pragma unroll
  for (int e = 0; e < 16; ++e) {
    const int kk = kbase + e + hi * 16;
    b[e] = (__bf16)wcol[(size_t)kk * wstride];
  }
  return b;
}

// ---- Repack weights [1024, ncols] fp32 -> bf16 B-fragments.
// Fragment f = nb*KBLKS + kb; lane l elem e: K = kb*32 + (l>>4)*16 + e,
// N = nb*16 + (l&15).
__global__ __launch_bounds__(256)
void pack_weights_kernel(const float* __restrict__ src, __bf16* __restrict__ dst,
                         int ncols, int total_dwords) {
  const int gid = blockIdx.x * 256 + threadIdx.x;
  if (gid >= total_dwords) return;
  const int r    = gid & 255;
  const int f    = gid >> 8;
  const int lane = r & 31;
  const int j    = r >> 5;          // elems 2j, 2j+1
  const int nb = f >> 5;
  const int kb = f & 31;
  const int n = nb * 16 + (lane & 15);
  const int k = kb * KTILE + (lane >> 4) * 16 + 2 * j;
  union { __bf16 h[2]; unsigned u; } pk;
  pk.h[0] = (__bf16)src[(size_t)k * ncols + n];
  pk.h[1] = (__bf16)src[(size_t)(k + 1) * ncols + n];
  *(unsigned*)(dst + (size_t)f * FRAG_ELEMS + lane * 16 + 2 * j) = pk.u;
}

// ---- Repack activations [rows, 1024] fp32 -> bf16 A-fragments.
// Coalesced float2 reads along K; scattered dword writes into fragments.
__global__ __launch_bounds__(256)
void pack_acts_kernel(const float* __restrict__ src, __bf16* __restrict__ dst,
                      int total_pairs) {
  const int gid = blockIdx.x * 256 + threadIdx.x;
  if (gid >= total_pairs) return;
  const int row = gid >> 9;          // / (HDIM/2)
  const int k   = (gid & 511) * 2;
  const float2 f = *(const float2*)(src + (size_t)row * HDIM + k);
  const int kb = k >> 5;
  const int kk = k & 31;
  const int hi = kk >> 4;                              // half-wave
  const int e  = kk & 15;                              // elem (even)
  const int lane = hi * 16 + (row & 15);
  const int mb = row >> 4;
  union { __bf16 h[2]; unsigned u; } pk;
  pk.h[0] = (__bf16)f.x;
  pk.h[1] = (__bf16)f.y;
  *(unsigned*)(dst + (((size_t)mb * KBLKS + kb) << 9) + lane * 16 + e) = pk.u;
}

template <bool APACK, bool BPACK>
__global__ __launch_bounds__(256)
void lstm_layer_kernel(const float*  __restrict__ x_f,     // [B,1024] fp32
                       const float*  __restrict__ hprev_f, // [B,1024] fp32
                       const __bf16* __restrict__ x_p,     // packed A frags
                       const __bf16* __restrict__ hprev_p,
                       const float*  __restrict__ c_prev,  // [B,1024]
                       const float*  __restrict__ Wi_f,    // [1024,4096]
                       const float*  __restrict__ Wh_f,
                       const __bf16* __restrict__ Wi_p,    // packed B frags
                       const __bf16* __restrict__ Wh_p,
                       const float*  __restrict__ bi,      // [4096]
                       const float*  __restrict__ bh,
                       float* __restrict__ h_out,
                       float* __restrict__ c_out)
{
  const int wave = threadIdx.x >> 5;
  const int lane = threadIdx.x & 31;
  const int lo   = lane & 15;
  const int hi   = lane >> 4;

  // Block tile 64 rows x 64 h-cols; wave tile 32 rows x 16 cols.
  const int m0  = blockIdx.x * 64 + (wave >> 2) * 32;
  const int n0  = blockIdx.y * 64 + (wave & 3) * 16;
  const int nb0 = n0 >> 4;
  const int mb0 = m0 >> 4;
  const int col = n0 + lo;

  v8f acc[4][2];
#pragma unroll
  for (int g = 0; g < 4; ++g) {
    const float bias = bi[g * HDIM + col] + bh[g * HDIM + col];
#pragma unroll
    for (int sm = 0; sm < 2; ++sm)
#pragma unroll
      for (int j = 0; j < 8; ++j) acc[g][sm][j] = bias;
  }

#pragma unroll
  for (int phase = 0; phase < 2; ++phase) {
    const float*  __restrict__ af = phase ? hprev_f : x_f;
    const __bf16* __restrict__ ap = phase ? hprev_p : x_p;
    const float*  __restrict__ wf = phase ? Wh_f : Wi_f;
    const __bf16* __restrict__ wp = phase ? Wh_p : Wi_p;

    const float* arow0 = nullptr;
    const float* arow1 = nullptr;
    if constexpr (!APACK) {
      arow0 = af + (size_t)(m0 + lo) * HDIM;
      arow1 = arow0 + (size_t)16 * HDIM;
    }

    for (int kb = 0; kb < KBLKS; ++kb) {
      v16bf a0, a1;
      if constexpr (APACK) {
        a0 = load_frag_packed(ap, (size_t)mb0 * KBLKS + kb, lane);
        a1 = load_frag_packed(ap, (size_t)(mb0 + 1) * KBLKS + kb, lane);
      } else {
        const int k0 = kb * KTILE;
        a0 = load_a_frag(arow0, k0, hi);
        a1 = load_a_frag(arow1, k0, hi);
      }
#pragma unroll
      for (int g = 0; g < 4; ++g) {
        v16bf b;
        if constexpr (BPACK) {
          b = load_frag_packed(wp, (size_t)(g * (HDIM / 16) + nb0) * KBLKS + kb,
                               lane);
        } else {
          b = load_b_gather(wf + (g * HDIM + col), kb * KTILE, hi, GSTRIDE);
        }
        acc[g][0] = __builtin_amdgcn_wmma_f32_16x16x32_bf16(
            false, a0, false, b, (short)0, acc[g][0], false, false);
        acc[g][1] = __builtin_amdgcn_wmma_f32_16x16x32_bf16(
            false, a1, false, b, (short)0, acc[g][1], false, false);
      }
    }
  }

  // Cell update. C/D layout: VGPR j -> row = m0 + sm*16 + hi*8 + j, col n0+lo.
#pragma unroll
  for (int sm = 0; sm < 2; ++sm) {
#pragma unroll
    for (int j = 0; j < 8; ++j) {
      const int r = m0 + sm * 16 + hi * 8 + j;
      const size_t idx = (size_t)r * HDIM + col;
      const float ig = fast_sigmoid(acc[0][sm][j]);
      const float fg = fast_sigmoid(acc[1][sm][j]);
      const float gg = fast_tanh(acc[2][sm][j]);
      const float og = fast_sigmoid(acc[3][sm][j]);
      const float c  = fg * c_prev[idx] + ig * gg;
      const float h  = og * fast_tanh(c);
      c_out[idx] = c;
      h_out[idx] = h;
    }
  }
}

template <bool APACK, bool BPACK>
__global__ __launch_bounds__(256)
void fc_kernel(const float*  __restrict__ hlast_f,  // [B,1024] fp32
               const __bf16* __restrict__ hlast_p,  // packed A frags
               const float*  __restrict__ W_f,      // [1024,1024]
               const __bf16* __restrict__ W_p,      // packed B frags
               const float*  __restrict__ bias,     // [1024]
               float* __restrict__ y)               // [B,1024]
{
  const int wave = threadIdx.x >> 5;
  const int lane = threadIdx.x & 31;
  const int lo   = lane & 15;
  const int hi   = lane >> 4;

  const int m0  = blockIdx.x * 64 + (wave >> 2) * 32;
  const int n0  = blockIdx.y * 64 + (wave & 3) * 16;
  const int nb0 = n0 >> 4;
  const int mb0 = m0 >> 4;
  const int col = n0 + lo;

  v8f acc[2];
  {
    const float b = bias[col];
#pragma unroll
    for (int sm = 0; sm < 2; ++sm)
#pragma unroll
      for (int j = 0; j < 8; ++j) acc[sm][j] = b;
  }

  const float* arow0 = nullptr;
  const float* arow1 = nullptr;
  if constexpr (!APACK) {
    arow0 = hlast_f + (size_t)(m0 + lo) * HDIM;
    arow1 = arow0 + (size_t)16 * HDIM;
  }

  for (int kb = 0; kb < KBLKS; ++kb) {
    v16bf a0, a1;
    if constexpr (APACK) {
      a0 = load_frag_packed(hlast_p, (size_t)mb0 * KBLKS + kb, lane);
      a1 = load_frag_packed(hlast_p, (size_t)(mb0 + 1) * KBLKS + kb, lane);
    } else {
      const int k0 = kb * KTILE;
      a0 = load_a_frag(arow0, k0, hi);
      a1 = load_a_frag(arow1, k0, hi);
    }
    v16bf b;
    if constexpr (BPACK) {
      b = load_frag_packed(W_p, (size_t)nb0 * KBLKS + kb, lane);
    } else {
      b = load_b_gather(W_f + col, kb * KTILE, hi, HDIM);
    }
    acc[0] = __builtin_amdgcn_wmma_f32_16x16x32_bf16(
        false, a0, false, b, (short)0, acc[0], false, false);
    acc[1] = __builtin_amdgcn_wmma_f32_16x16x32_bf16(
        false, a1, false, b, (short)0, acc[1], false, false);
  }

#pragma unroll
  for (int sm = 0; sm < 2; ++sm)
#pragma unroll
    for (int j = 0; j < 8; ++j) {
      const int r = m0 + sm * 16 + hi * 8 + j;
      y[(size_t)r * HDIM + col] = acc[sm][j];
    }
}

extern "C" void kernel_launch(void* const* d_in, const int* in_sizes, int n_in,
                              void* d_out, int out_size, void* d_ws, size_t ws_size,
                              hipStream_t stream) {
  (void)in_sizes; (void)n_in; (void)out_size;

  const float* x      = (const float*)d_in[0];
  const float* h_prev = (const float*)d_in[1];
  const float* c_prev = (const float*)d_in[2];
  const float* Wi     = (const float*)d_in[3];
  const float* Wh     = (const float*)d_in[4];
  const float* bi     = (const float*)d_in[5];
  const float* bh     = (const float*)d_in[6];
  const float* fcW    = (const float*)d_in[7];
  const float* fcb    = (const float*)d_in[8];

  float* out   = (float*)d_out;
  const size_t nBH = (size_t)BATCH * HDIM;
  float* y     = out;
  float* hbase = out + nBH;
  float* cbase = hbase + (size_t)LAYERS * nBH;

  const size_t wElems   = (size_t)HDIM * GSTRIDE;   // one Wi/Wh matrix
  const size_t fcElems  = (size_t)HDIM * HDIM;
  const size_t wTotal   = LAYERS * 2 * wElems + fcElems;   // packed weight elems
  const size_t wBytes   = wTotal * 2;
  const size_t actBytes = 2 * nBH * 2;              // two bf16 act buffers

  const int tier = (d_ws && ws_size >= wBytes + actBytes) ? 2
                 : (d_ws && ws_size >= wBytes)            ? 1
                                                          : 0;

  const dim3 blk(256);
  const dim3 grid(BATCH / 64, HDIM / 64);
  const int wDwords  = (int)(wElems / 2);
  const int wPGrid   = (wDwords + 255) / 256;
  const int fcDwords = (int)(fcElems / 2);
  const int aPairs   = (int)(nBH / 2);
  const int aPGrid   = (aPairs + 255) / 256;

  __bf16* wsb  = (__bf16*)d_ws;
  __bf16* actA = wsb + wTotal;        // layer input activations (packed)
  __bf16* actH = actA + nBH;          // layer h_prev activations (packed)
  __bf16* fcp  = wsb + (size_t)(2 * LAYERS) * wElems;

  if (tier >= 1) {
    for (int l = 0; l < LAYERS; ++l) {
      pack_weights_kernel<<<wPGrid, blk, 0, stream>>>(
          Wi + (size_t)l * wElems, wsb + (size_t)(2 * l) * wElems,
          GSTRIDE, wDwords);
      pack_weights_kernel<<<wPGrid, blk, 0, stream>>>(
          Wh + (size_t)l * wElems, wsb + (size_t)(2 * l + 1) * wElems,
          GSTRIDE, wDwords);
    }
    pack_weights_kernel<<<(fcDwords + 255) / 256, blk, 0, stream>>>(
        fcW, fcp, HDIM, fcDwords);
  }

  if (tier == 2) {
    pack_acts_kernel<<<aPGrid, blk, 0, stream>>>(x, actA, aPairs);
    for (int l = 0; l < LAYERS; ++l) {
      float* h_out = hbase + (size_t)l * nBH;
      float* c_out = cbase + (size_t)l * nBH;
      pack_acts_kernel<<<aPGrid, blk, 0, stream>>>(
          h_prev + (size_t)l * nBH, actH, aPairs);
      lstm_layer_kernel<true, true><<<grid, blk, 0, stream>>>(
          nullptr, nullptr, actA, actH,
          c_prev + (size_t)l * nBH,
          nullptr, nullptr,
          wsb + (size_t)(2 * l) * wElems, wsb + (size_t)(2 * l + 1) * wElems,
          bi + (size_t)l * GSTRIDE, bh + (size_t)l * GSTRIDE,
          h_out, c_out);
      pack_acts_kernel<<<aPGrid, blk, 0, stream>>>(h_out, actA, aPairs);
    }
    fc_kernel<true, true><<<grid, blk, 0, stream>>>(
        nullptr, actA, nullptr, fcp, fcb, y);
  } else if (tier == 1) {
    const float* x_in = x;
    for (int l = 0; l < LAYERS; ++l) {
      float* h_out = hbase + (size_t)l * nBH;
      float* c_out = cbase + (size_t)l * nBH;
      lstm_layer_kernel<false, true><<<grid, blk, 0, stream>>>(
          x_in, h_prev + (size_t)l * nBH, nullptr, nullptr,
          c_prev + (size_t)l * nBH,
          nullptr, nullptr,
          wsb + (size_t)(2 * l) * wElems, wsb + (size_t)(2 * l + 1) * wElems,
          bi + (size_t)l * GSTRIDE, bh + (size_t)l * GSTRIDE,
          h_out, c_out);
      x_in = h_out;
    }
    fc_kernel<false, true><<<grid, blk, 0, stream>>>(
        hbase + (size_t)(LAYERS - 1) * nBH, nullptr, nullptr, fcp, fcb, y);
  } else {
    const float* x_in = x;
    for (int l = 0; l < LAYERS; ++l) {
      float* h_out = hbase + (size_t)l * nBH;
      float* c_out = cbase + (size_t)l * nBH;
      lstm_layer_kernel<false, false><<<grid, blk, 0, stream>>>(
          x_in, h_prev + (size_t)l * nBH, nullptr, nullptr,
          c_prev + (size_t)l * nBH,
          Wi + (size_t)l * wElems, Wh + (size_t)l * wElems, nullptr, nullptr,
          bi + (size_t)l * GSTRIDE, bh + (size_t)l * GSTRIDE,
          h_out, c_out);
      x_in = h_out;
    }
    fc_kernel<false, false><<<grid, blk, 0, stream>>>(
        hbase + (size_t)(LAYERS - 1) * nBH, nullptr, fcW, nullptr, fcb, y);
  }
}
